// TransformerBlock_29308856828543
// MI455X (gfx1250) — matrix operand order repaired and checked by reading, not compiled, not run on hardware
//
#include <hip/hip_runtime.h>

typedef unsigned short u16;
typedef unsigned int   u32;
typedef __attribute__((ext_vector_type(16))) __bf16 v16bf;
typedef __attribute__((ext_vector_type(8)))  float  v8f;

#define B_   16
#define NTOK 4096
#define C_   256
#define MROW 65536   // B_ * NTOK
#define HID  1024
#define HGT  64
#define WID  64

__device__ __forceinline__ u16 f2bf(float f) {
  u32 u = __float_as_uint(f);
  u += 0x7FFFu + ((u >> 16) & 1u);          // round-to-nearest-even
  return (u16)(u >> 16);
}
__device__ __forceinline__ float bf2f(u16 h) { return __uint_as_float(((u32)h) << 16); }
__device__ __forceinline__ float gelu_f(float v) { return 0.5f * v * (1.0f + erff(v * 0.70710678f)); }
__device__ __forceinline__ float sigmoid_f(float v) { return 1.0f / (1.0f + expf(-v)); }

// ---------------- weight f32 -> bf16 ----------------
__global__ void cvt_kernel(const float* __restrict__ s, u16* __restrict__ d, int n) {
  int i = blockIdx.x * 256 + threadIdx.x;
  if (i < n) d[i] = f2bf(s[i]);
}

// ---------------- LayerNorm over channels (one block = one token) ----------------
__global__ void __launch_bounds__(256) ln_kernel(const float* __restrict__ in,
    const float* __restrict__ g, const float* __restrict__ bta,
    float* __restrict__ outF, u16* __restrict__ outB) {
  int tok = blockIdx.x, c = threadIdx.x;
  float v = in[(size_t)tok * C_ + c];
  __shared__ float r1[256], r2[256];
  r1[c] = v; r2[c] = v * v; __syncthreads();
  for (int s = 128; s > 0; s >>= 1) {
    if (c < s) { r1[c] += r1[c + s]; r2[c] += r2[c + s]; }
    __syncthreads();
  }
  float mean = r1[0] * (1.0f / C_);
  float var  = r2[0] * (1.0f / C_) - mean * mean;
  float rs = rsqrtf(var + 1e-5f);
  float o = (v - mean) * rs * g[c] + bta[c];
  if (outF) outF[(size_t)tok * C_ + c] = o;
  if (outB) outB[(size_t)tok * C_ + c] = f2bf(o);
}

// ---------------- dw3x3(sr) + LN(an) + GELU + add xn -> xa (bf16) ----------------
__global__ void __launch_bounds__(256) sr_kernel(const float* __restrict__ xn,
    const float* __restrict__ sw, const float* __restrict__ sb,
    const float* __restrict__ ag, const float* __restrict__ ab,
    u16* __restrict__ xa) {
  int tok = blockIdx.x, c = threadIdx.x;
  int b = tok >> 12, n = tok & (NTOK - 1);
  int y = n >> 6, x = n & 63;
  float acc = sb[c];
  for (int ky = 0; ky < 3; ky++) {
    int yy = y + ky - 1; if (yy < 0 || yy >= HGT) continue;
    for (int kx = 0; kx < 3; kx++) {
      int xx = x + kx - 1; if (xx < 0 || xx >= WID) continue;
      acc += xn[((size_t)((b << 12) + (yy << 6) + xx)) * C_ + c] * sw[c * 9 + ky * 3 + kx];
    }
  }
  __shared__ float r1[256], r2[256];
  r1[c] = acc; r2[c] = acc * acc; __syncthreads();
  for (int s = 128; s > 0; s >>= 1) {
    if (c < s) { r1[c] += r1[c + s]; r2[c] += r2[c + s]; }
    __syncthreads();
  }
  float mean = r1[0] * (1.0f / C_);
  float var  = r2[0] * (1.0f / C_) - mean * mean;
  float rs = rsqrtf(var + 1e-5f);
  float t = (acc - mean) * rs * ag[c] + ab[c];
  xa[(size_t)tok * C_ + c] = f2bf(xn[(size_t)tok * C_ + c] + gelu_f(t));
}

// ---------------- bf16 WMMA GEMM: out[m,n] = sum_k A[m,k]*W[n,k] (+bias +res1 +res2) ---
// Roles swapped: WMMA A-frag <- W tile (n x k), B-frag <- X tile (k x m), D[n][m].
// Block tile: 128(M) x 128(N); 8 waves, each wave: 16(n) x 128(m) = 8 WMMAs per K-step,
// one A-fragment reused across all 8 B-fragments.
#define SXS 40
#define SWS 40
__global__ void __launch_bounds__(256) gemm_kernel(
    const u16* __restrict__ A, const u16* __restrict__ W,
    int M, int N, int K,
    float* __restrict__ outF, u16* __restrict__ outB,
    const float* __restrict__ bias,
    const float* __restrict__ res1, const float* __restrict__ res2) {
  __shared__ __align__(16) u16 sX[128 * SXS];   // activation tile, 128 rows(m) x 32(k)
  __shared__ __align__(16) u16 sW[128 * SWS];   // weight tile, 128 rows(n) x 32(k)
  int tid = threadIdx.x, lane = tid & 31, wave = tid >> 5;
  int mBase = blockIdx.x * 128, nBase = blockIdx.y * 128;
  int laneLo = lane & 15;
  int kA0 = (lane & 16) ? 8 : 0;    // A-frag: elems 0..7 -> k=kA0.., elems 8..15 -> k=kA0+16..
  int kB0 = (lane & 16) ? 8 : 0;    // B-frag: elems 0..7 -> k=kB0.., elems 8..15 -> k=kB0+16..
  int nLoc = wave * 16 + laneLo;
  v8f acc[8] = {};
  union UV { v16bf v; uint4 q[2]; };
  for (int k0 = 0; k0 < K; k0 += 32) {
    for (int i = 0; i < 2; i++) { // load 128x32 activation tile
      int l = tid + i * 256;
      int r = l >> 2, c8 = (l & 3) << 3;
      *(uint4*)&sX[r * SXS + c8] = *(const uint4*)(A + (size_t)(mBase + r) * K + k0 + c8);
    }
    for (int i = 0; i < 2; i++) { // load 128x32 weight tile
      int l = tid + i * 256;
      int r = l >> 2, c8 = (l & 3) << 3;
      *(uint4*)&sW[r * SWS + c8] = *(const uint4*)(W + (size_t)(nBase + r) * K + k0 + c8);
    }
    __syncthreads();
    UV ua;
    ua.q[0] = *(const uint4*)&sW[nLoc * SWS + kA0];
    ua.q[1] = *(const uint4*)&sW[nLoc * SWS + kA0 + 16];
#pragma unroll
    for (int mf = 0; mf < 8; mf++) {
      UV ub;
      int mLoc = mf * 16 + laneLo;
      ub.q[0] = *(const uint4*)&sX[mLoc * SXS + kB0];
      ub.q[1] = *(const uint4*)&sX[mLoc * SXS + kB0 + 16];
      acc[mf] = __builtin_amdgcn_wmma_f32_16x16x32_bf16(
          false, ua.v, false, ub.v, (short)0, acc[mf], false, false);
    }
    __syncthreads();
  }
  // D element r (v8f): n = nBase + wave*16 + (lane<16?0:8) + r ; m = mBase + mf*16 + (lane&15)
  int nOff = nBase + wave * 16 + ((lane & 16) ? 8 : 0);
  float bv[8];
#pragma unroll
  for (int r = 0; r < 8; r++) bv[r] = bias ? bias[nOff + r] : 0.0f;
#pragma unroll
  for (int mf = 0; mf < 8; mf++) {
    int m = mBase + mf * 16 + laneLo;
    size_t base = (size_t)m * N + nOff;
    float v[8];
#pragma unroll
    for (int r = 0; r < 8; r++) v[r] = acc[mf][r] + bv[r];
    if (res1) {
#pragma unroll
      for (int r = 0; r < 8; r++) v[r] += res1[base + r];
    }
    if (res2) {
#pragma unroll
      for (int r = 0; r < 8; r++) v[r] += res2[base + r];
    }
    if (outF) {
      float4 lo = make_float4(v[0], v[1], v[2], v[3]);
      float4 hi = make_float4(v[4], v[5], v[6], v[7]);
      *(float4*)(outF + base) = lo;
      *(float4*)(outF + base + 4) = hi;
    } else {
      uint4 q;
      q.x = (u32)f2bf(v[0]) | ((u32)f2bf(v[1]) << 16);
      q.y = (u32)f2bf(v[2]) | ((u32)f2bf(v[3]) << 16);
      q.z = (u32)f2bf(v[4]) | ((u32)f2bf(v[5]) << 16);
      q.w = (u32)f2bf(v[6]) | ((u32)f2bf(v[7]) << 16);
      *(uint4*)(outB + base) = q;
    }
  }
}

// ---------------- online column softmax stats for q: per (b,h,d) over N tokens -------
__global__ void qstats_kernel(const float* __restrict__ qkv,
                              float* __restrict__ qmax, float* __restrict__ qsum) {
  int d = threadIdx.x, tg = threadIdx.y;
  int bh = blockIdx.x, b = bh >> 3, h = bh & 7;
  const float* qp = qkv + ((size_t)b * NTOK) * 768 + h * 32 + d;
  float m = -3.4e38f, s = 0.0f;
  for (int n = tg; n < NTOK; n += 8) {
    float v = qp[(size_t)n * 768];
    float nm = fmaxf(m, v);
    s = s * expf(m - nm) + expf(v - nm);
    m = nm;
  }
  __shared__ float sm[8][32], ss[8][32];
  sm[tg][d] = m; ss[tg][d] = s; __syncthreads();
  if (tg == 0) {
    for (int i = 1; i < 8; i++) {
      float mi = sm[i][d], si = ss[i][d];
      float nm = fmaxf(m, mi);
      s = s * expf(m - nm) + si * expf(mi - nm);
      m = nm;
    }
    qmax[bh * 32 + d] = m;
    qsum[bh * 32 + d] = s;
  }
}

// ---------------- ctx = softmax_row(k)^T @ v : 32x32 per (b,h), k row-softmax fused --
__global__ void __launch_bounds__(256) ctx_kernel(const float* __restrict__ qkv,
                                                  float* __restrict__ ctxg) {
  int bh = blockIdx.x, b = bh >> 3, h = bh & 7;
  int lane = threadIdx.x & 31, wave = threadIdx.x >> 5;   // wave = token-in-chunk, lane = dim
  const float* kbase = qkv + ((size_t)b * NTOK) * 768 + 256 + h * 32;
  const float* vbase = qkv + ((size_t)b * NTOK) * 768 + 512 + h * 32;
  __shared__ float kb[8][32], vb[8][32];
  float acc[4] = {0.f, 0.f, 0.f, 0.f};
  for (int n0 = 0; n0 < NTOK; n0 += 8) {
    int n = n0 + wave;
    float kv = kbase[(size_t)n * 768 + lane];
    float vv = vbase[(size_t)n * 768 + lane];
    float mx = kv;
    for (int o = 16; o > 0; o >>= 1) mx = fmaxf(mx, __shfl_xor(mx, o, 32));
    float e = expf(kv - mx);
    float sm = e;
    for (int o = 16; o > 0; o >>= 1) sm += __shfl_xor(sm, o, 32);
    kb[wave][lane] = e / sm;
    vb[wave][lane] = vv;
    __syncthreads();
#pragma unroll
    for (int t = 0; t < 8; t++) {
      float kk = kb[t][lane];
#pragma unroll
      for (int j = 0; j < 4; j++) acc[j] += kk * vb[t][wave * 4 + j];
    }
    __syncthreads();
  }
#pragma unroll
  for (int j = 0; j < 4; j++)
    ctxg[(size_t)bh * 1024 + lane * 32 + wave * 4 + j] = acc[j];   // ctx[d][e]
}

// ---------------- attn[n,e] = sum_d softmaxcol(q)[n,d] * ctx[d,e] -> bf16 [B,N,C] ----
__global__ void __launch_bounds__(256) attn_kernel(const float* __restrict__ qkv,
    const float* __restrict__ ctxg, const float* __restrict__ qmax,
    const float* __restrict__ qsum, u16* __restrict__ attnb) {
  int bh = blockIdx.x, b = bh >> 3, h = bh & 7;
  int tid = threadIdx.x, t = tid >> 5, e = tid & 31;
  __shared__ float cx[32][32];
  __shared__ float qm[32], qi[32];
  __shared__ float ql[8][32];
  for (int i = 0; i < 4; i++) {
    int idx = tid + i * 256;
    cx[idx >> 5][idx & 31] = ctxg[(size_t)bh * 1024 + idx];
  }
  if (tid < 32) { qm[tid] = qmax[bh * 32 + tid]; qi[tid] = 1.0f / qsum[bh * 32 + tid]; }
  __syncthreads();
  for (int it = 0; it < 16; it++) {
    int n = blockIdx.y * 128 + it * 8 + t;
    float qraw = qkv[((size_t)(b * NTOK + n)) * 768 + h * 32 + e];
    ql[t][e] = expf(qraw - qm[e]) * qi[e];
    __syncthreads();
    float s = 0.0f;
#pragma unroll
    for (int d2 = 0; d2 < 32; d2++) s += ql[t][d2] * cx[d2][e];
    attnb[((size_t)(b * NTOK + n)) * C_ + h * 32 + e] = f2bf(s);
    __syncthreads();
  }
}

// ---------------- MixFFN dwconv 3x3 (HID channels) + GELU: y1(bf16) -> y2(bf16) ------
__global__ void __launch_bounds__(256) mix_kernel(const u16* __restrict__ y1,
    const float* __restrict__ dww, const float* __restrict__ dwb, u16* __restrict__ y2) {
  int tok = blockIdx.x;
  int b = tok >> 12, n = tok & (NTOK - 1);
  int y = n >> 6, x = n & 63;
  for (int j = 0; j < 4; j++) {
    int c = threadIdx.x + j * 256;
    float acc = dwb[c];
    for (int ky = 0; ky < 3; ky++) {
      int yy = y + ky - 1; if (yy < 0 || yy >= HGT) continue;
      for (int kx = 0; kx < 3; kx++) {
        int xx = x + kx - 1; if (xx < 0 || xx >= WID) continue;
        acc += bf2f(y1[((size_t)((b << 12) + (yy << 6) + xx)) * HID + c]) * dww[c * 9 + ky * 3 + kx];
      }
    }
    y2[(size_t)tok * HID + c] = f2bf(gelu_f(acc));
  }
}

// ---------------- channel pools: per (b,c) mean & max over N tokens ------------------
__global__ void __launch_bounds__(1024) pool_kernel(const float* __restrict__ x3,
    float* __restrict__ pavg, float* __restrict__ pmx) {
  int b = blockIdx.x, c = threadIdx.x & 255, tg = threadIdx.x >> 8;
  const float* p = x3 + ((size_t)b * NTOK) * C_ + c;
  float s = 0.0f, m = -3.4e38f;
  for (int n = tg; n < NTOK; n += 4) { float v = p[(size_t)n * C_]; s += v; m = fmaxf(m, v); }
  __shared__ float ls[4][256], lm[4][256];
  ls[tg][c] = s; lm[tg][c] = m; __syncthreads();
  if (tg == 0) {
    for (int i = 1; i < 4; i++) { s += ls[i][c]; m = fmaxf(m, lm[i][c]); }
    pavg[b * C_ + c] = s * (1.0f / NTOK);
    pmx[b * C_ + c] = m;
  }
}

// ---------------- channel attention MLP (256->16->256, shared weights, sum, sigmoid) --
__global__ void __launch_bounds__(256) ca_kernel(const float* __restrict__ pavg,
    const float* __restrict__ pmx, const float* __restrict__ w1,
    const float* __restrict__ w2, float* __restrict__ cab) {
  int b = blockIdx.x, c = threadIdx.x;
  __shared__ float sa[256], sm[256], h1[32];
  sa[c] = pavg[b * C_ + c]; sm[c] = pmx[b * C_ + c]; __syncthreads();
  if (c < 32) {
    int j = c & 15;
    const float* src = (c < 16) ? sa : sm;
    float s = 0.0f;
    for (int k = 0; k < 256; k++) s += src[k] * w1[j * 256 + k];
    h1[c] = fmaxf(s, 0.0f);
  }
  __syncthreads();
  float s = 0.0f;
#pragma unroll
  for (int j = 0; j < 16; j++) s += (h1[j] + h1[16 + j]) * w2[c * 16 + j];
  cab[b * C_ + c] = sigmoid_f(s);
}

// ---------------- spatial map: per token mean & max over channels of x3*ca ------------
__global__ void __launch_bounds__(256) smap_kernel(const float* __restrict__ x3,
    const float* __restrict__ cab, float* __restrict__ smean, float* __restrict__ smax) {
  int b = blockIdx.x >> 9;                 // 512 blocks per batch image
  __shared__ float cal[256];
  cal[threadIdx.x] = cab[b * C_ + threadIdx.x];
  __syncthreads();
  int t = threadIdx.x >> 5, lane = threadIdx.x & 31;
  int idx = blockIdx.x * 8 + t;            // global token
  const float* row = x3 + (size_t)idx * C_;
  float s = 0.0f, m = -3.4e38f;
#pragma unroll
  for (int j = 0; j < 8; j++) {
    float v = row[lane * 8 + j] * cal[lane * 8 + j];
    s += v; m = fmaxf(m, v);
  }
  for (int o = 16; o > 0; o >>= 1) { s += __shfl_xor(s, o, 32); m = fmaxf(m, __shfl_xor(m, o, 32)); }
  if (lane == 0) { smean[idx] = s * (1.0f / C_); smax[idx] = m; }
}

// ---------------- 7x7 spatial conv over [mean,max] + sigmoid -> sa[b,n] --------------
__global__ void spconv_kernel(const float* __restrict__ smean, const float* __restrict__ smax,
    const float* __restrict__ spw, const float* __restrict__ spb, float* __restrict__ sab) {
  int i = blockIdx.x * 256 + threadIdx.x;  // [0, MROW)
  int b = i >> 12, n = i & (NTOK - 1);
  int y = n >> 6, x = n & 63;
  float acc = spb[0];
  for (int ky = 0; ky < 7; ky++) {
    int yy = y + ky - 3; if (yy < 0 || yy >= HGT) continue;
    for (int kx = 0; kx < 7; kx++) {
      int xx = x + kx - 3; if (xx < 0 || xx >= WID) continue;
      int nn = (b << 12) + (yy << 6) + xx;
      acc += smean[nn] * spw[ky * 7 + kx] + smax[nn] * spw[49 + ky * 7 + kx];
    }
  }
  sab[i] = sigmoid_f(acc);
}

// ---------------- final: out = x3 * ca * sa (in-place on d_out) ----------------------
__global__ void final_kernel(const float* __restrict__ x3, const float* __restrict__ cab,
                             const float* __restrict__ sab, float* __restrict__ out) {
  size_t i = (size_t)blockIdx.x * 256 + threadIdx.x;   // over MROW*C_
  int c = (int)(i & 255);
  int tok = (int)(i >> 8);
  int b = tok >> 12;
  out[i] = x3[i] * cab[b * C_ + c] * sab[tok];
}

extern "C" void kernel_launch(void* const* d_in, const int* in_sizes, int n_in,
                              void* d_out, int out_size, void* d_ws, size_t ws_size,
                              hipStream_t stream) {
  (void)in_sizes; (void)n_in; (void)out_size; (void)ws_size;
  const float* x      = (const float*)d_in[0];
  const float* n1_g   = (const float*)d_in[3];
  const float* n1_b   = (const float*)d_in[4];
  const float* sr_w   = (const float*)d_in[5];
  const float* sr_b   = (const float*)d_in[6];
  const float* an_g   = (const float*)d_in[7];
  const float* an_b   = (const float*)d_in[8];
  const float* qkv_w  = (const float*)d_in[9];
  const float* proj_w = (const float*)d_in[10];
  const float* proj_b = (const float*)d_in[11];
  const float* n2_g   = (const float*)d_in[12];
  const float* n2_b   = (const float*)d_in[13];
  const float* fc1_w  = (const float*)d_in[14];
  const float* fc1_b  = (const float*)d_in[15];
  const float* dw_w   = (const float*)d_in[16];
  const float* dw_b   = (const float*)d_in[17];
  const float* fc2_w  = (const float*)d_in[18];
  const float* fc2_b  = (const float*)d_in[19];
  const float* ca_w1  = (const float*)d_in[20];
  const float* ca_w2  = (const float*)d_in[21];
  const float* sp_w   = (const float*)d_in[22];
  const float* sp_b   = (const float*)d_in[23];

  float* x23 = (float*)d_out;               // x2 then x3 live in d_out (fully rewritten)

  char* ws = (char*)d_ws;
  const size_t MB = (size_t)1 << 20;
  // arena with liveness-based reuse (peak ~355MB)
  float* xn    = (float*)(ws);              // 64MB  (dead after proj)
  u16*   xa    = (u16*)(ws + 64 * MB);      // 32MB  (dead after qkv gemm)
  float* qkv   = (float*)(ws + 96 * MB);    // 192MB (dead after attn kernel)
  u16*   attnb = (u16*)(ws + 64 * MB);      // 32MB  (reuses xa)
  u16*   xn2   = (u16*)(ws);                // 32MB  (reuses xn)
  u16*   y1    = (u16*)(ws + 96 * MB);      // 128MB (reuses qkv)
  u16*   y2    = (u16*)(ws + 224 * MB);     // 128MB
  char*  misc  = ws + 352 * MB;
  u16*   wqkv  = (u16*)(misc + 0);          // 768*256 bf16
  u16*   wproj = (u16*)(misc + 393216);     // 256*256 bf16
  u16*   wfc1  = (u16*)(misc + 524288);     // 1024*256 bf16
  u16*   wfc2  = (u16*)(misc + 1048576);    // 256*1024 bf16
  float* qmax  = (float*)(misc + 1572864);
  float* qsum  = (float*)(misc + 1589248);
  float* ctxg  = (float*)(misc + 1605632);  // 128*1024 f32
  float* pavg  = (float*)(misc + 2129920);
  float* pmx   = (float*)(misc + 2146304);
  float* cab   = (float*)(misc + 2162688);
  float* smean = (float*)(misc + 2179072);
  float* smaxb = (float*)(misc + 2441216);
  float* sab   = (float*)(misc + 2703360);

  // weights -> bf16 (recomputed every call: deterministic)
  cvt_kernel<<<(768 * 256 + 255) / 256, 256, 0, stream>>>(qkv_w, wqkv, 768 * 256);
  cvt_kernel<<<(256 * 256 + 255) / 256, 256, 0, stream>>>(proj_w, wproj, 256 * 256);
  cvt_kernel<<<(1024 * 256 + 255) / 256, 256, 0, stream>>>(fc1_w, wfc1, 1024 * 256);
  cvt_kernel<<<(256 * 1024 + 255) / 256, 256, 0, stream>>>(fc2_w, wfc2, 256 * 1024);

  // attention path
  ln_kernel<<<MROW, 256, 0, stream>>>(x, n1_g, n1_b, xn, nullptr);
  sr_kernel<<<MROW, 256, 0, stream>>>(xn, sr_w, sr_b, an_g, an_b, xa);
  gemm_kernel<<<dim3(MROW / 128, 768 / 128), 256, 0, stream>>>(
      xa, wqkv, MROW, 768, 256, qkv, nullptr, nullptr, nullptr, nullptr);
  qstats_kernel<<<128, dim3(32, 8), 0, stream>>>(qkv, qmax, qsum);
  ctx_kernel<<<128, 256, 0, stream>>>(qkv, ctxg);
  attn_kernel<<<dim3(128, 32), 256, 0, stream>>>(qkv, ctxg, qmax, qsum, attnb);
  gemm_kernel<<<dim3(MROW / 128, 256 / 128), 256, 0, stream>>>(
      attnb, wproj, MROW, 256, 256, x23, nullptr, proj_b, x, xn);

  // MixFFN path
  ln_kernel<<<MROW, 256, 0, stream>>>(x23, n2_g, n2_b, nullptr, xn2);
  gemm_kernel<<<dim3(MROW / 128, 1024 / 128), 256, 0, stream>>>(
      xn2, wfc1, MROW, 1024, 256, nullptr, y1, fc1_b, nullptr, nullptr);
  mix_kernel<<<MROW, 256, 0, stream>>>(y1, dw_w, dw_b, y2);
  gemm_kernel<<<dim3(MROW / 128, 256 / 128), 256, 0, stream>>>(
      y2, wfc2, MROW, 256, 1024, x23, nullptr, fc2_b, x23, nullptr);

  // CSDA path
  pool_kernel<<<B_, 1024, 0, stream>>>(x23, pavg, pmx);
  ca_kernel<<<B_, 256, 0, stream>>>(pavg, pmx, ca_w1, ca_w2, cab);
  smap_kernel<<<MROW / 8, 256, 0, stream>>>(x23, cab, smean, smaxb);
  spconv_kernel<<<MROW / 256, 256, 0, stream>>>(smean, smaxb, sp_w, sp_b, sab);
  final_kernel<<<MROW, 256, 0, stream>>>(x23, cab, sab, x23);
}
